// Transformer_1468878815263
// MI455X (gfx1250) — matrix operand order repaired and checked by reading, not compiled, not run on hardware
//
#include <hip/hip_runtime.h>

typedef __attribute__((ext_vector_type(16))) _Float16 v16h;
typedef __attribute__((ext_vector_type(8)))  _Float16 v8h;
typedef __attribute__((ext_vector_type(8)))  float    v8f;
using half_t = _Float16;

#define B_     2
#define T_     1024
#define META_  6
#define T1_    1030        // T + META
#define D_     512
#define H_     8
#define HS_    64
#define L_     6
#define V_     32000
#define CH_    6
#define MROWS  (B_ * T1_)  // 2060
#define MPAD   2112        // 33 * 64
#define LOG1E4 9.210340371976184f

// ---------------------------------------------------------------------------
// Weight packing: fp32 [K,N] (possibly strided per head) -> f16 [N,K] row-major
// ---------------------------------------------------------------------------
__global__ __launch_bounds__(256) void k_pack_qkv(
    const float* __restrict__ Wq, const float* __restrict__ Wk,
    const float* __restrict__ Wv, half_t* __restrict__ dst) {
  // dst layout: [L][1536][512]; n in [0,512)=Q, [512,1024)=K, [1024,1536)=V
  long total = (long)L_ * 1536 * 512;
  long i = (long)blockIdx.x * 256 + threadIdx.x;
  if (i >= total) return;
  int k = (int)(i % 512);
  int n = (int)((i / 512) % 1536);
  int l = (int)(i / (512 * 1536));
  int seg = n / 512;
  int hh  = (n % 512) / HS_;
  int e   = n % HS_;
  const float* W = (seg == 0) ? Wq : (seg == 1) ? Wk : Wv;
  dst[i] = (half_t)W[(((size_t)l * H_ + hh) * D_ + k) * HS_ + e];
}

__global__ __launch_bounds__(256) void k_pack_T(
    const float* __restrict__ src, half_t* __restrict__ dst, int K, int N, int L) {
  long total = (long)L * K * N;
  long i = (long)blockIdx.x * 256 + threadIdx.x;
  if (i >= total) return;
  int k = (int)(i % K);
  int n = (int)((i / K) % N);
  int l = (int)(i / ((long)K * N));
  dst[i] = (half_t)src[((size_t)l * K + k) * N + n];
}

// ---------------------------------------------------------------------------
// Embedding + chunked positional encoding
// ---------------------------------------------------------------------------
__global__ __launch_bounds__(256) void k_embed(
    const int* __restrict__ idx, const int* __restrict__ midx,
    const float* __restrict__ tok, const float* __restrict__ met,
    float* __restrict__ x) {
  int r = blockIdx.x;             // [0, MROWS)
  int b = r / T1_, t = r - b * T1_;
  for (int d = threadIdx.x; d < D_; d += 256) {
    float v;
    if (t < T_) {
      int tk = idx[b * T_ + t];
      float p  = (float)(t / CH_);
      int   m  = d >> 1;
      float dv = __expf((float)(2 * m) * (-LOG1E4 / (float)D_));
      float ang = p * dv;
      float pe = (d & 1) ? __cosf(ang) : __sinf(ang);
      v = tok[(size_t)tk * D_ + d] + pe;
    } else {
      int mi = midx[b * META_ + (t - T_)];
      v = met[(size_t)mi * D_ + d];
    }
    x[(size_t)r * D_ + d] = v;
  }
}

// ---------------------------------------------------------------------------
// LayerNorm over D=512, fp32 in -> f16 out
// ---------------------------------------------------------------------------
__global__ __launch_bounds__(256) void k_ln(
    const float* __restrict__ X, const float* __restrict__ g,
    const float* __restrict__ bb, half_t* __restrict__ O) {
  __shared__ float red[256];
  int r = blockIdx.x, tid = threadIdx.x;
  const float* xr = X + (size_t)r * D_;
  float a0 = xr[tid], a1 = xr[tid + 256];
  red[tid] = a0 + a1;
  __syncthreads();
  for (int s = 128; s > 0; s >>= 1) { if (tid < s) red[tid] += red[tid + s]; __syncthreads(); }
  float mu = red[0] * (1.0f / D_);
  __syncthreads();
  float d0 = a0 - mu, d1 = a1 - mu;
  red[tid] = d0 * d0 + d1 * d1;
  __syncthreads();
  for (int s = 128; s > 0; s >>= 1) { if (tid < s) red[tid] += red[tid + s]; __syncthreads(); }
  float rs = rsqrtf(red[0] * (1.0f / D_) + 1e-5f);
  half_t* orow = O + (size_t)r * D_;
  orow[tid]       = (half_t)(d0 * rs * g[tid]       + bb[tid]);
  orow[tid + 256] = (half_t)(d1 * rs * g[tid + 256] + bb[tid + 256]);
}

// ---------------------------------------------------------------------------
// WMMA fragment loads (row-major A [M,K], pre-transposed B as Bt [N,K])
// A 16x32 f16 layout: lanes 0-15 hold K kk+0..7 (v0..3) and kk+16..23 (v4..7);
//                     lanes 16-31 hold kk+8..15 and kk+24..31.
// B 32x16 f16 layout: same K order as A (lanes 0-15 kk+0..7, kk+16..23; lanes 16-31 kk+8..15, kk+24..31).
// ---------------------------------------------------------------------------
__device__ __forceinline__ v16h ldfragA(const half_t* p) {  // p = &Arow[kk + hs*8]
  union { v16h v; struct { v8h lo, hi; } s; } u;
  u.s.lo = *(const v8h*)p;
  u.s.hi = *(const v8h*)(p + 16);
  return u.v;
}
__device__ __forceinline__ v16h ldfragB(const half_t* p) {  // p = &Bcol[kk + hs*8]
  union { v16h v; struct { v8h lo, hi; } s; } u;
  u.s.lo = *(const v8h*)p;
  u.s.hi = *(const v8h*)(p + 16);
  return u.v;
}

struct FragSet {
  v16h a[4];
  v16h b0, b1;
};

__device__ __forceinline__ void load_set(FragSet& f, const half_t* const ap[4],
                                         const half_t* bp0, const half_t* bp1,
                                         int kk) {
#pragma unroll
  for (int m = 0; m < 4; ++m) f.a[m] = ldfragA(ap[m] + kk);
  f.b0 = ldfragB(bp0 + kk);
  f.b1 = ldfragB(bp1 + kk);
}

__device__ __forceinline__ void mma_set(v8f acc[4][2], const FragSet& f) {
#pragma unroll
  for (int m = 0; m < 4; ++m) {
    acc[m][0] = __builtin_amdgcn_wmma_f32_16x16x32_f16(
        false, f.a[m], false, f.b0, (short)0, acc[m][0], false, false);
    acc[m][1] = __builtin_amdgcn_wmma_f32_16x16x32_f16(
        false, f.a[m], false, f.b1, (short)0, acc[m][1], false, false);
  }
}

__device__ __forceinline__ void epi_store(float v, int row, int col, int N, int flags,
                                          float* __restrict__ Cf, half_t* __restrict__ Ch,
                                          const float* __restrict__ resid,
                                          float bval) {
  v += bval;
  if (flags & 4) v += resid[(size_t)row * N + col];
  if (flags & 2) v = fmaxf(v, 0.0f);
  if (flags & 16) {                                   // LM head: [b, t<T, V] remap
    int b = row / T1_, t = row - b * T1_;
    if (row < MROWS && t < T_) Cf[((size_t)b * T_ + t) * V_ + col] = v;
  } else if (flags & 8) {
    Ch[(size_t)row * N + col] = (half_t)v;
  } else {
    Cf[(size_t)row * N + col] = v;
  }
}

// ---------------------------------------------------------------------------
// WMMA GEMM: C[M,N] = A_f16[M,K] * Bt_f16[N,K]^T (+bias, +resid, relu, remap)
// block = 128 threads (4 waves); block tile 64x128; wave tile 64x32
// (4 M-subtiles x 2 N-subtiles = 8 accumulators).
// K-loop unrolled x2 with ping-pong register buffers (s0/s1): no cur<-next
// register copies, loads for the next 32-K chunk always in flight under the
// WMMAs of the current chunk. Requires K % 64 == 0 (holds: 512, 2048).
// flags: 1=bias  2=relu  4=+resid(fp32, ld=N)  8=store f16  16=LM-head remap
// ---------------------------------------------------------------------------
__global__ __launch_bounds__(128) void k_gemm(
    const half_t* __restrict__ A, const half_t* __restrict__ Bt,
    float* __restrict__ Cf, half_t* __restrict__ Ch,
    const float* __restrict__ bias, const float* __restrict__ resid,
    int N, int K, int flags) {
  const int lane = threadIdx.x & 31;
  const int wave = threadIdx.x >> 5;
  const int l16  = lane & 15;
  const int hs   = (lane >> 4) & 1;
  const int rowBase = blockIdx.y * 64;
  const int col0 = blockIdx.x * 128 + wave * 32 + l16;
  const int col1 = col0 + 16;

  const half_t* ap[4];
#pragma unroll
  for (int m = 0; m < 4; ++m)
    ap[m] = A + (size_t)(rowBase + m * 16 + l16) * K + hs * 8;
  const half_t* bp0 = Bt + (size_t)col0 * K + hs * 8;
  const half_t* bp1 = Bt + (size_t)col1 * K + hs * 8;

  v8f acc[4][2] = {};
  FragSet s0, s1;

  load_set(s0, ap, bp0, bp1, 0);
  int kk = 0;
  for (; kk + 64 < K; kk += 64) {
    load_set(s1, ap, bp0, bp1, kk + 32);
    __builtin_prefetch(bp0 + kk + 64, 0, 1);   // global_prefetch_b8
    __builtin_prefetch(bp1 + kk + 64, 0, 1);
    mma_set(acc, s0);
    load_set(s0, ap, bp0, bp1, kk + 64);
    mma_set(acc, s1);
  }
  // peeled final 64-K chunk: no more loads after s1
  load_set(s1, ap, bp0, bp1, kk + 32);
  mma_set(acc, s0);
  mma_set(acc, s1);

  const float bv0 = (flags & 1) ? bias[col0] : 0.0f;
  const float bv1 = (flags & 1) ? bias[col1] : 0.0f;
#pragma unroll
  for (int m = 0; m < 4; ++m) {
    int rbase = rowBase + m * 16 + (hs << 3);   // C layout: lanes>=16 -> M+8
#pragma unroll
    for (int i = 0; i < 8; ++i) {
      int row = rbase + i;
      epi_store(acc[m][0][i], row, col0, N, flags, Cf, Ch, resid, bv0);
      epi_store(acc[m][1][i], row, col1, N, flags, Cf, Ch, resid, bv1);
    }
  }
}

// ---------------------------------------------------------------------------
// Attention: one block per (b, h, query i). Chunk-6 causal + metadata mask.
// qkv fp32 [MPAD,1536]: cols [0,512)=Q, [512,1024)=K, [1024,1536)=V.
// ---------------------------------------------------------------------------
__global__ __launch_bounds__(128) void k_attn(
    const float* __restrict__ qkv, half_t* __restrict__ o) {
  __shared__ float sc[T1_];
  __shared__ float qs[HS_];
  __shared__ float red[128];
  const int tid = threadIdx.x;
  const int gi  = blockIdx.x;
  const int i   = gi % T1_;
  const int bh  = gi / T1_;
  const int h   = bh % H_;
  const int b   = bh / H_;
  const float scale = 0.044194173824159216f;       // 512^-0.5 (full D, per ref)

  const float* qp = qkv + ((size_t)(b * T1_ + i) * 1536) + h * HS_;
  if (tid < HS_) qs[tid] = qp[tid];
  __syncthreads();

  const int limit = (i / CH_ + 1) * CH_;
  float lmax = -INFINITY;
  for (int j = tid; j < T1_; j += 128) {
    float sv;
    if (j < limit || j >= T1_ - CH_) {
      const float* kp = qkv + ((size_t)(b * T1_ + j) * 1536) + 512 + h * HS_;
      float d = 0.0f;
#pragma unroll
      for (int e = 0; e < HS_; ++e) d += qs[e] * kp[e];
      sv = d * scale;
    } else {
      sv = -INFINITY;
    }
    sc[j] = sv;
    lmax = fmaxf(lmax, sv);
  }
  red[tid] = lmax;
  __syncthreads();
  for (int s = 64; s > 0; s >>= 1) { if (tid < s) red[tid] = fmaxf(red[tid], red[tid + s]); __syncthreads(); }
  float mx = red[0];
  __syncthreads();

  float lsum = 0.0f;
  for (int j = tid; j < T1_; j += 128) {
    float p = __expf(sc[j] - mx);   // masked -> exp(-inf) = 0
    sc[j] = p;
    lsum += p;
  }
  red[tid] = lsum;
  __syncthreads();
  for (int s = 64; s > 0; s >>= 1) { if (tid < s) red[tid] += red[tid + s]; __syncthreads(); }
  float inv = 1.0f / red[0];

  if (tid < HS_) {
    const float* vbase = qkv + 1024 + h * HS_ + tid;
    float acc = 0.0f;
    for (int j = 0; j < T1_; ++j)
      acc += sc[j] * vbase[(size_t)(b * T1_ + j) * 1536];
    o[((size_t)(b * T1_ + i) * D_) + h * HS_ + tid] = (half_t)(acc * inv);
  }
}

// ---------------------------------------------------------------------------
// Orchestration
// ---------------------------------------------------------------------------
extern "C" void kernel_launch(void* const* d_in, const int* in_sizes, int n_in,
                              void* d_out, int out_size, void* d_ws, size_t ws_size,
                              hipStream_t stream) {
  const int*   idx   = (const int*)  d_in[0];
  const int*   midx  = (const int*)  d_in[1];
  const float* tok   = (const float*)d_in[2];
  const float* met   = (const float*)d_in[3];
  const float* Wq    = (const float*)d_in[4];
  const float* Wk    = (const float*)d_in[5];
  const float* Wv    = (const float*)d_in[6];
  const float* Wo    = (const float*)d_in[7];
  const float* bo    = (const float*)d_in[8];
  const float* W1    = (const float*)d_in[9];
  const float* b1    = (const float*)d_in[10];
  const float* W2    = (const float*)d_in[11];
  const float* b2    = (const float*)d_in[12];
  const float* ln1g  = (const float*)d_in[13];
  const float* ln1b  = (const float*)d_in[14];
  const float* ln2g  = (const float*)d_in[15];
  const float* ln2b  = (const float*)d_in[16];
  const float* lnfg  = (const float*)d_in[17];
  const float* lnfb  = (const float*)d_in[18];
  const float* Wlm   = (const float*)d_in[19];
  const float* blm   = (const float*)d_in[20];
  float* out = (float*)d_out;

  // --- workspace carve-up (256B aligned) ---
  char* w = (char*)d_ws;
  size_t off = 0;
  auto carve = [&](size_t bytes) {
    size_t o = off;
    off = (off + bytes + 255) & ~(size_t)255;
    return o;
  };
  half_t* qkvT = (half_t*)(w + carve((size_t)L_ * 1536 * 512 * 2));
  half_t* WoT  = (half_t*)(w + carve((size_t)L_ * 512  * 512 * 2));
  half_t* W1T  = (half_t*)(w + carve((size_t)L_ * 2048 * 512 * 2));
  half_t* W2T  = (half_t*)(w + carve((size_t)L_ * 512  * 2048 * 2));
  half_t* WlmT = (half_t*)(w + carve((size_t)V_ * 512 * 2));
  float*  x    = (float*) (w + carve((size_t)MPAD * 512  * 4));
  half_t* hbuf = (half_t*)(w + carve((size_t)MPAD * 512  * 2));
  float*  qkv  = (float*) (w + carve((size_t)MPAD * 1536 * 4));
  half_t* obuf = (half_t*)(w + carve((size_t)MPAD * 512  * 2));
  half_t* hid  = (half_t*)(w + carve((size_t)MPAD * 2048 * 2));
  half_t* xf   = (half_t*)(w + carve((size_t)MPAD * 512  * 2));

  // --- pack weights to f16 [N,K] ---
  {
    long n;
    n = (long)L_ * 1536 * 512;
    k_pack_qkv<<<dim3((unsigned)((n + 255) / 256)), 256, 0, stream>>>(Wq, Wk, Wv, qkvT);
    n = (long)L_ * 512 * 512;
    k_pack_T<<<dim3((unsigned)((n + 255) / 256)), 256, 0, stream>>>(Wo, WoT, 512, 512, L_);
    n = (long)L_ * 512 * 2048;
    k_pack_T<<<dim3((unsigned)((n + 255) / 256)), 256, 0, stream>>>(W1, W1T, 512, 2048, L_);
    n = (long)L_ * 2048 * 512;
    k_pack_T<<<dim3((unsigned)((n + 255) / 256)), 256, 0, stream>>>(W2, W2T, 2048, 512, L_);
    n = (long)V_ * 512;
    k_pack_T<<<dim3((unsigned)((n + 255) / 256)), 256, 0, stream>>>(Wlm, WlmT, 512, V_, 1);
  }

  // --- embedding ---
  k_embed<<<MROWS, 256, 0, stream>>>(idx, midx, tok, met, x);

  const dim3 gemm_blk(128);
  const unsigned mtiles = MPAD / 64;   // 33

  for (int l = 0; l < L_; ++l) {
    // LN1 -> h (f16)
    k_ln<<<MROWS, 256, 0, stream>>>(x, ln1g + l * D_, ln1b + l * D_, hbuf);
    // QKV: [MPAD,512] x [512,1536] -> qkv fp32
    k_gemm<<<dim3(1536 / 128, mtiles), gemm_blk, 0, stream>>>(
        hbuf, qkvT + (size_t)l * 1536 * 512, qkv, nullptr, nullptr, nullptr,
        1536, 512, 0);
    // attention -> o (f16)
    k_attn<<<B_ * H_ * T1_, 128, 0, stream>>>(qkv, obuf);
    // out proj + bo + residual -> x (fp32, in place)
    k_gemm<<<dim3(512 / 128, mtiles), gemm_blk, 0, stream>>>(
        obuf, WoT + (size_t)l * 512 * 512, x, nullptr, bo + l * D_, x,
        512, 512, 1 | 4);
    // LN2 -> h
    k_ln<<<MROWS, 256, 0, stream>>>(x, ln2g + l * D_, ln2b + l * D_, hbuf);
    // MLP1 + b1 + relu -> hid (f16)
    k_gemm<<<dim3(2048 / 128, mtiles), gemm_blk, 0, stream>>>(
        hbuf, W1T + (size_t)l * 2048 * 512, nullptr, hid, b1 + l * 2048, nullptr,
        2048, 512, 1 | 2 | 8);
    // MLP2 + b2 + residual -> x
    k_gemm<<<dim3(512 / 128, mtiles), gemm_blk, 0, stream>>>(
        hid, W2T + (size_t)l * 512 * 2048, x, nullptr, b2 + l * D_, x,
        512, 2048, 1 | 4);
  }

  // final LN -> xf ; LM head -> d_out with [b, t<T, V] remap
  k_ln<<<MROWS, 256, 0, stream>>>(x, lnfg, lnfb, xf);
  k_gemm<<<dim3(V_ / 128, mtiles), gemm_blk, 0, stream>>>(
      xf, WlmT, out, nullptr, blm, nullptr, V_, 512, 1 | 16);
}
